// Encoder_53592601919961
// MI455X (gfx1250) — matrix operand order repaired and checked by reading, not compiled, not run on hardware
//
#include <hip/hip_runtime.h>
#include <hip/hip_bf16.h>

// B=256, TEND=256, D_IN=256, H=256, 4H=1024
#define Bsz 256
#define Tn  256
#define Dn  256
#define Hn  256
#define NBLK 64          // persistent grid: 4 row-groups x 16 col-groups
#define NWAVES_PER_BLK 4
#define ARRIVALS (NBLK * NWAVES_PER_BLK)   // 256 wave-arrivals per step

typedef _Float16 v8h  __attribute__((ext_vector_type(8)));
typedef _Float16 v16h __attribute__((ext_vector_type(16)));
typedef float    v8f  __attribute__((ext_vector_type(8)));

__device__ __forceinline__ float sigmoidf_(float x) { return 1.0f / (1.0f + __expf(-x)); }

// A fragment (16x32 f16, MxK): lane<16 holds K {kb+0..7, kb+16..23}; lane>=16 holds {kb+8..15, kb+24..31}
__device__ __forceinline__ v16h load_frag_a(const _Float16* p) {
    v8h lo = *(const v8h*)(p);
    v8h hi = *(const v8h*)(p + 16);
    return __builtin_shufflevector(lo, hi, 0,1,2,3,4,5,6,7,8,9,10,11,12,13,14,15);
}
// Pre-swizzled fragment: 16 contiguous f16 at this lane's slot (LDS or global)
__device__ __forceinline__ v16h load_frag_16(const _Float16* p) {
    v8h lo = *(const v8h*)(p);
    v8h hi = *(const v8h*)(p + 8);
    return __builtin_shufflevector(lo, hi, 0,1,2,3,4,5,6,7,8,9,10,11,12,13,14,15);
}

// ---------------- Kernel 0: weight conversion + state/counter init ----------------
__global__ __launch_bounds__(256) void prep_weights(
    const float* __restrict__ W_ih, const float* __restrict__ W_hh,
    const float* __restrict__ b_ih, const float* __restrict__ b_hh,
    _Float16* __restrict__ Wih_h, _Float16* __restrict__ Whh_h,
    float* __restrict__ bsum, _Float16* __restrict__ hbuf, int* __restrict__ ctr)
{
    int i = blockIdx.x * blockDim.x + threadIdx.x;       // grid covers 4H*Dn = 262144
    if (i < 4 * Hn * Dn) {
        Wih_h[i] = (_Float16)W_ih[i];
        Whh_h[i] = (_Float16)W_hh[i];
    }
    if (i < 4 * Hn) bsum[i] = b_ih[i] + b_hh[i];
    if (i < 2 * Bsz * Hn) hbuf[i] = (_Float16)0.0f;      // both ping-pong halves
    if (i < Tn) ctr[i] = 0;                              // per-step arrival counters
}

// ---------------- Kernel 1: x_term + softmax (time-invariant attention) ----------------
// score = (h.w_h + c.w_c + b)[:,None] + x_term : per-row constant cancels in softmax,
// so a = softmax(x_term) is independent of t/h/c -> compute once.
__global__ __launch_bounds__(256) void attn_softmax(
    const float* __restrict__ data, const float* __restrict__ attn_W,
    float* __restrict__ a_out)
{
    const int b = blockIdx.x;
    const int d = threadIdx.x;
    const float* wx = attn_W + 2 * Hn;                   // w_x = attn_W[0][2H:]
    const float* p  = data + (size_t)b * Tn * Dn + d;
    float s = 0.0f;
    #pragma unroll 4
    for (int t = 0; t < Tn; ++t) s += p[(size_t)t * Dn] * wx[t];

    __shared__ float red[256];
    red[d] = s; __syncthreads();
    for (int off = 128; off > 0; off >>= 1) {
        if (d < off) red[d] = fmaxf(red[d], red[d + off]);
        __syncthreads();
    }
    float mx = red[0]; __syncthreads();
    float e = __expf(s - mx);
    red[d] = e; __syncthreads();
    for (int off = 128; off > 0; off >>= 1) {
        if (d < off) red[d] += red[d + off];
        __syncthreads();
    }
    a_out[b * Dn + d] = e * (1.0f / red[0]);
}

// ---------------- Kernel 2: wi[b,t,d] = tanh(a[b,d]*x[b,t,d])  (all timesteps, parallel) ---
__global__ __launch_bounds__(256) void compute_wi(
    const float* __restrict__ data, const float* __restrict__ a_in,
    float* __restrict__ out_wi /* [B][T][D] */, _Float16* __restrict__ WI_h /* [T][B][D] */)
{
    const int b = blockIdx.x >> 8;
    const int t = blockIdx.x & 255;
    const int d = threadIdx.x;
    float x = data[((size_t)b * Tn + t) * Dn + d];
    float w = tanhf(a_in[b * Dn + d] * x);
    out_wi[((size_t)b * Tn + t) * Dn + d] = w;
    WI_h[((size_t)t * Bsz + b) * Dn + d] = (_Float16)w;
}

// ---------------- Kernel 3: persistent recurrence over all 256 steps -------------------
// 64 blocks x 128 threads; wave -> one 16-row x 16-col patch, all four gate tiles.
// Weights for this block's column group are staged ONCE into LDS, pre-swizzled into the
// per-lane WMMA B-fragment order (chunk c = mat*32+g*8+j is 1KB; lane owns bytes l*32..+31).
// Steps are chained with a device-scope release/acquire arrival counter (one signal per
// wave, spin with s_sleep); h ping-pongs between two buffers so one barrier/step suffices.
// Cell state c stays in registers for the entire sequence.
__global__ __launch_bounds__(128) void lstm_persistent(
    const _Float16* __restrict__ WI_h,  // [T][B][D] f16
    const _Float16* __restrict__ Wih_h, // [4H][D]   f16
    const _Float16* __restrict__ Whh_h, // [4H][H]   f16
    const float*    __restrict__ bsum,  // [4H]
    _Float16*       __restrict__ hbuf,  // [2][B][H] f16 ping-pong
    float*          __restrict__ out_h, // [B][T][H] f32 (input_encoded)
    int*            __restrict__ ctr)   // [T] arrival counters (zeroed by prep)
{
    extern __shared__ _Float16 Wlds[];          // 64 chunks * 512 f16 = 64KB

    const int wave    = threadIdx.x >> 5;       // 0..3
    const int lane    = threadIdx.x & 31;
    const int cg      = blockIdx.x & 15;        // column group (16 H-cols)
    const int rg      = blockIdx.x >> 4;        // row group (64 rows)
    const int rowbase = (rg * 4 + wave) * 16;
    const int colbase = cg * 16;
    const int m       = lane & 15;
    const bool laneLo = (lane < 16);
    const int aoff    = laneLo ? 0 : 8;
    const int boff    = laneLo ? 0 : 8;

    // ---- stage this column group's weights into LDS, pre-swizzled ----
    for (int c = wave; c < 64; c += NWAVES_PER_BLK) {
        const int mat = c >> 5;                 // 0: W_ih, 1: W_hh
        const int g   = (c >> 3) & 3;           // gate
        const int j   = c & 7;                  // k-block
        const _Float16* src = (mat ? Whh_h : Wih_h)
                            + (size_t)(colbase + g * Hn + m) * Dn + j * 32 + boff;
        v8h lo = *(const v8h*)(src);
        v8h hi = *(const v8h*)(src + 16);
        _Float16* dst = Wlds + c * 512 + lane * 16;
        *(v8h*)(dst)     = lo;
        *(v8h*)(dst + 8) = hi;
    }
    __syncthreads();

    const float bi = bsum[0 * Hn + colbase + m];
    const float bf = bsum[1 * Hn + colbase + m];
    const float bg = bsum[2 * Hn + colbase + m];
    const float bo = bsum[3 * Hn + colbase + m];
    const int radd = laneLo ? 0 : 8;

    float cr[8];
    #pragma unroll
    for (int r = 0; r < 8; ++r) cr[r] = 0.0f;   // cell state lives in registers

    const _Float16* Arow = WI_h + (size_t)(rowbase + m) * Dn + aoff;   // += B*D per step
    const _Float16* Wih_l = Wlds + lane * 16;           // chunks 0..31
    const _Float16* Whh_l = Wlds + 32 * 512 + lane * 16; // chunks 32..63

    for (int t = 0; t < Tn; ++t) {
        v8f acc0 = {}, acc1 = {}, acc2 = {}, acc3 = {};

        // ---- phase 1: wi_t @ W_ih^T (h-independent -> overlaps the step barrier) ----
        #pragma unroll
        for (int j = 0; j < 8; ++j) {
            v16h a  = load_frag_a(Arow + j * 32);
            v16h b0 = load_frag_16(Wih_l + (0 * 8 + j) * 512);
            v16h b1 = load_frag_16(Wih_l + (1 * 8 + j) * 512);
            v16h b2 = load_frag_16(Wih_l + (2 * 8 + j) * 512);
            v16h b3 = load_frag_16(Wih_l + (3 * 8 + j) * 512);
            acc0 = __builtin_amdgcn_wmma_f32_16x16x32_f16(false, a, false, b0, (short)0, acc0, false, false);
            acc1 = __builtin_amdgcn_wmma_f32_16x16x32_f16(false, a, false, b1, (short)0, acc1, false, false);
            acc2 = __builtin_amdgcn_wmma_f32_16x16x32_f16(false, a, false, b2, (short)0, acc2, false, false);
            acc3 = __builtin_amdgcn_wmma_f32_16x16x32_f16(false, a, false, b3, (short)0, acc3, false, false);
        }

        // ---- wait until h_{t-1} from all blocks is visible ----
        if (t > 0) {
            const int* p = ctr + (t - 1);
            while (__hip_atomic_load(p, __ATOMIC_ACQUIRE, __HIP_MEMORY_SCOPE_AGENT) < ARRIVALS)
                __builtin_amdgcn_s_sleep(1);
        }
        const _Float16* Hrow = hbuf + (size_t)((t & 1) ? Bsz * Hn : 0)
                             + (size_t)(rowbase + m) * Hn + aoff;

        // ---- phase 2: h_{t-1} @ W_hh^T ----
        #pragma unroll
        for (int j = 0; j < 8; ++j) {
            v16h a  = load_frag_a(Hrow + j * 32);
            v16h b0 = load_frag_16(Whh_l + (0 * 8 + j) * 512);
            v16h b1 = load_frag_16(Whh_l + (1 * 8 + j) * 512);
            v16h b2 = load_frag_16(Whh_l + (2 * 8 + j) * 512);
            v16h b3 = load_frag_16(Whh_l + (3 * 8 + j) * 512);
            acc0 = __builtin_amdgcn_wmma_f32_16x16x32_f16(false, a, false, b0, (short)0, acc0, false, false);
            acc1 = __builtin_amdgcn_wmma_f32_16x16x32_f16(false, a, false, b1, (short)0, acc1, false, false);
            acc2 = __builtin_amdgcn_wmma_f32_16x16x32_f16(false, a, false, b2, (short)0, acc2, false, false);
            acc3 = __builtin_amdgcn_wmma_f32_16x16x32_f16(false, a, false, b3, (short)0, acc3, false, false);
        }

        // ---- LSTM cell update (D VGPR r -> M = r (+8 hi lanes), N = lane%16) ----
        _Float16* hw = hbuf + (size_t)(((t + 1) & 1) ? Bsz * Hn : 0);
        #pragma unroll
        for (int r = 0; r < 8; ++r) {
            const int brow = rowbase + r + radd;
            float iv = acc0[r] + bi;
            float fv = acc1[r] + bf;
            float gv = acc2[r] + bg;
            float ov = acc3[r] + bo;
            float cnew = sigmoidf_(fv) * cr[r] + sigmoidf_(iv) * tanhf(gv);
            float hnew = sigmoidf_(ov) * tanhf(cnew);
            cr[r] = cnew;
            hw[(size_t)brow * Hn + colbase + m] = (_Float16)hnew;
            out_h[((size_t)brow * Tn + t) * Hn + colbase + m] = hnew;
        }

        // ---- signal arrival (release orders this wave's h stores) ----
        if (lane == 0)
            __hip_atomic_fetch_add(ctr + t, 1, __ATOMIC_RELEASE, __HIP_MEMORY_SCOPE_AGENT);

        Arow += (size_t)Bsz * Dn;
    }
}

extern "C" void kernel_launch(void* const* d_in, const int* in_sizes, int n_in,
                              void* d_out, int out_size, void* d_ws, size_t ws_size,
                              hipStream_t stream) {
    const float* data   = (const float*)d_in[0];
    const float* attn_W = (const float*)d_in[1];
    // d_in[2] = attn_b: per-row constant cancels inside the softmax -> unused
    const float* W_ih   = (const float*)d_in[3];
    const float* W_hh   = (const float*)d_in[4];
    const float* b_ih   = (const float*)d_in[5];
    const float* b_hh   = (const float*)d_in[6];

    float* out_wi = (float*)d_out;                                  // [B][T][D]
    float* out_h  = (float*)d_out + (size_t)Bsz * Tn * Dn;          // [B][T][H]

    // workspace layout (~34.7 MB), all chunks 16B-aligned
    char* ws = (char*)d_ws;
    float*    a_buf = (float*)ws;                       // B*D      f32  (256 KB)
    float*    bsum  = a_buf + Bsz * Dn;                 // 4H       f32  (4 KB)
    _Float16* Wih_h = (_Float16*)(bsum + 4 * Hn);       // 4H*D     f16  (512 KB)
    _Float16* Whh_h = Wih_h + 4 * Hn * Dn;              // 4H*H     f16  (512 KB)
    _Float16* hbuf  = Whh_h + 4 * Hn * Hn;              // 2*B*H    f16  (256 KB)
    int*      ctr   = (int*)(hbuf + 2 * Bsz * Hn);      // T        i32  (1 KB)
    _Float16* WI_h  = (_Float16*)(ctr + Tn);            // T*B*D    f16  (33.5 MB)

    prep_weights<<<(4 * Hn * Dn) / 256, 256, 0, stream>>>(W_ih, W_hh, b_ih, b_hh,
                                                          Wih_h, Whh_h, bsum, hbuf, ctr);
    attn_softmax<<<Bsz, 256, 0, stream>>>(data, attn_W, a_buf);
    compute_wi<<<Bsz * Tn, 256, 0, stream>>>(data, a_buf, out_wi, WI_h);

    // one persistent launch replaces 256 per-step launches
    lstm_persistent<<<NBLK, NWAVES_PER_BLK * 32, 64 * 512 * sizeof(_Float16), stream>>>(
        WI_h, Wih_h, Whh_h, bsum, hbuf, out_h, ctr);
}
